// KANExpert_9156870275586
// MI455X (gfx1250) — matrix operand order repaired and checked by reading, not compiled, not run on hardware
//
#include <hip/hip_runtime.h>

typedef _Float16 half8 __attribute__((ext_vector_type(8)));
typedef _Float16 v16h  __attribute__((ext_vector_type(16)));
typedef float    v8f   __attribute__((ext_vector_type(8)));

#define NOUT 256
#define NB   (NOUT / 16)     // 16 N-blocks of 16 columns
#define BM   128             // rows per workgroup
#define NTHR 256             // 8 wave32

// ---------------------------------------------------------------------------
// Cubic B-spline bases (GRID_SIZE=5, SPLINE_ORDER=3, uniform knots, h=0.4).
// Mirrors the reference Cox–de Boor recursion; returns 8 f16 values.
// ---------------------------------------------------------------------------
__device__ __forceinline__ half8 bspline8(float x) {
    const float h = 0.4f;
    float g[12];
#pragma unroll
    for (int j = 0; j < 12; ++j) g[j] = (float)(j - 3) * h - 1.0f;
    float b[11];
#pragma unroll
    for (int j = 0; j < 11; ++j)
        b[j] = (x >= g[j] && x < g[j + 1]) ? 1.0f : 0.0f;
#pragma unroll
    for (int k = 1; k <= 3; ++k) {
        const float inv = 1.0f / ((float)k * h);
#pragma unroll
        for (int j = 0; j + k < 11; ++j)
            b[j] = ((x - g[j]) * b[j] + (g[j + k + 1] - x) * b[j + 1]) * inv;
    }
    half8 o;
#pragma unroll
    for (int j = 0; j < 8; ++j) o[j] = (_Float16)b[j];
    return o;
}

// fast silu: result feeds an f16 operand, so raw v_rcp_f32 is plenty
__device__ __forceinline__ float silu(float x) {
    return x * __builtin_amdgcn_rcpf(1.0f + __expf(-x));
}

// ---------------------------------------------------------------------------
// Pack weights into WMMA-B fragment-major f16 layout.
// Block (kb,nb) covers K=kb*32..+31, N=nb*16..+15; 512 f16 per block:
//   K = kb*32 + (lane>=16 ? 16 : 0) + p*8 + e ;  N = nb*16 + (lane&15)
// so in the GEMM each lane reads its 16 operand halves as two coalesced b128s.
// ---------------------------------------------------------------------------
__global__ void pack_base_w(const float* __restrict__ W, _Float16* __restrict__ F, int IN) {
    int t = blockIdx.x * blockDim.x + threadIdx.x;
    int e  = t & 7;
    int l  = (t >> 3) & 31;
    int p  = (t >> 8) & 1;
    int nb = (t >> 9) & (NB - 1);
    int kb = t / (512 * NB);
    int k  = kb * 32 + ((l >> 4) << 4) + p * 8 + e;
    int n  = nb * 16 + (l & 15);
    F[t] = (_Float16)W[(size_t)n * IN + k];
}

__global__ void pack_spline_w(const float* __restrict__ W, const float* __restrict__ S,
                              _Float16* __restrict__ F, int IN) {
    int t = blockIdx.x * blockDim.x + threadIdx.x;
    int e  = t & 7;
    int l  = (t >> 3) & 31;
    int p  = (t >> 8) & 1;
    int nb = (t >> 9) & (NB - 1);
    int kb = t / (512 * NB);
    int k  = kb * 32 + ((l >> 4) << 4) + p * 8 + e;   // k = i*8 + g over K=8*IN
    int n  = nb * 16 + (l & 15);
    int i  = k >> 3;
    int gg = k & 7;
    F[t] = (_Float16)(W[((size_t)n * IN + i) * 8 + gg] * S[(size_t)n * IN + i]);
}

// ---------------------------------------------------------------------------
// One K-chunk of 32: A frags from LDS (ds_load_b128 pairs), B frags from
// packed global (coalesced b128 pairs), 16 WMMAs (4 M-frags x 4 N-frags).
// Wave w: wm = w&1 (64-row half), wn = w>>1 (64-col quarter).
// ---------------------------------------------------------------------------
__device__ __forceinline__ void mma_step(const _Float16 (*As)[40],
                                         const _Float16* __restrict__ wfrag,
                                         int kb, int nkb, int wm, int wn, int lane,
                                         v8f acc[4][4]) {
    const int arow = lane & 15;
    const int kOff = (lane >> 4) << 4;    // lanes 0-15: K 0..15 ; 16-31: 16..31
    v16h af[4];
#pragma unroll
    for (int m = 0; m < 4; ++m) {
        half8 lo = *(const half8*)&As[wm * 64 + m * 16 + arow][kOff];
        half8 hi = *(const half8*)&As[wm * 64 + m * 16 + arow][kOff + 8];
        af[m] = __builtin_shufflevector(lo, hi, 0,1,2,3,4,5,6,7,8,9,10,11,12,13,14,15);
    }
#pragma unroll
    for (int n = 0; n < 4; ++n) {
        const int nb = wn * 4 + n;
        const _Float16* blk = wfrag + (size_t)(kb * NB + nb) * 512;
        half8 lo = *(const half8*)(blk + lane * 8);
        half8 hi = *(const half8*)(blk + 256 + lane * 8);
        v16h bf = __builtin_shufflevector(lo, hi, 0,1,2,3,4,5,6,7,8,9,10,11,12,13,14,15);
        if (kb + 1 < nkb)
            __builtin_prefetch(blk + (size_t)NB * 512 + lane * 8, 0, 1);  // next K block
#pragma unroll
        for (int m = 0; m < 4; ++m)
            acc[m][n] = __builtin_amdgcn_wmma_f32_16x16x32_f16(
                false, af[m], false, bf, (short)0, acc[m][n], false, false);
    }
}

// ---------------------------------------------------------------------------
// Fused KAN layer: Out[B,256] = silu(X)@Wb^T + bases(X)@Ws^T, bases built
// on the fly into a double-buffered LDS f16 A-tile (never hits HBM).
// ---------------------------------------------------------------------------
__global__ __launch_bounds__(NTHR) void kan_layer_gemm(
    const float* __restrict__ X, const _Float16* __restrict__ wbase,
    const _Float16* __restrict__ wspl, float* __restrict__ Out, int IN) {
    __shared__ __align__(16) _Float16 As[2][BM][40];  // 80B rows: 16B aligned, bank-clean

    const int t    = threadIdx.x;
    const int lane = t & 31;
    const int wm   = (t >> 5) & 1;
    const int wn   = t >> 6;
    const int rowBase = blockIdx.x * BM;

    v8f acc[4][4] = {};

    // ---- phase 1: base path, A = silu(x), K = IN ----
    const int nkb1 = IN >> 5;
    {
        const int r   = t >> 1;          // 128 rows, 2 threads per row
        const int c16 = (t & 1) * 16;
        const float* xrow = X + (size_t)(rowBase + r) * IN + c16;
        // prologue fill of buffer 0
        {
            const float4* px = (const float4*)xrow;
            half8 h0, h1;
#pragma unroll
            for (int q = 0; q < 2; ++q) {
                float4 a = px[2 * q], b = px[2 * q + 1];
                half8& hv = q ? h1 : h0;
                hv[0] = (_Float16)silu(a.x); hv[1] = (_Float16)silu(a.y);
                hv[2] = (_Float16)silu(a.z); hv[3] = (_Float16)silu(a.w);
                hv[4] = (_Float16)silu(b.x); hv[5] = (_Float16)silu(b.y);
                hv[6] = (_Float16)silu(b.z); hv[7] = (_Float16)silu(b.w);
            }
            *(half8*)&As[0][r][c16] = h0;
            *(half8*)&As[0][r][c16 + 8] = h1;
        }
        __syncthreads();
        for (int kb = 0; kb < nkb1; ++kb) {
            const int cur = kb & 1;
            if (kb + 1 < nkb1) {                  // fill other buffer while MMAing
                const float4* px = (const float4*)(xrow + (kb + 1) * 32);
                half8 h0, h1;
#pragma unroll
                for (int q = 0; q < 2; ++q) {
                    float4 a = px[2 * q], b = px[2 * q + 1];
                    half8& hv = q ? h1 : h0;
                    hv[0] = (_Float16)silu(a.x); hv[1] = (_Float16)silu(a.y);
                    hv[2] = (_Float16)silu(a.z); hv[3] = (_Float16)silu(a.w);
                    hv[4] = (_Float16)silu(b.x); hv[5] = (_Float16)silu(b.y);
                    hv[6] = (_Float16)silu(b.z); hv[7] = (_Float16)silu(b.w);
                }
                *(half8*)&As[cur ^ 1][r][c16] = h0;
                *(half8*)&As[cur ^ 1][r][c16 + 8] = h1;
            }
            mma_step(As[cur], wbase, kb, nkb1, wm, wn, lane, acc);
            __syncthreads();
        }
    }

    // ---- phase 2: spline path, A = bases(x), K = 8*IN (chunk = 4 features) ----
    const int nkb2 = IN >> 2;
    {
        const int r0 = t >> 2;           // rows r0 and r0+64, 4 threads per row
        const int fi = t & 3;
        // prologue fill of buffer 0
#pragma unroll
        for (int s = 0; s < 2; ++s) {
            const int r = r0 + s * 64;
            float xv = X[(size_t)(rowBase + r) * IN + fi];
            *(half8*)&As[0][r][fi * 8] = bspline8(xv);
        }
        __syncthreads();
        for (int kb = 0; kb < nkb2; ++kb) {
            const int cur = kb & 1;
            if (kb + 1 < nkb2) {
#pragma unroll
                for (int s = 0; s < 2; ++s) {
                    const int r = r0 + s * 64;
                    float xv = X[(size_t)(rowBase + r) * IN + (kb + 1) * 4 + fi];
                    *(half8*)&As[cur ^ 1][r][fi * 8] = bspline8(xv);
                }
            }
            mma_step(As[cur], wspl, kb, nkb2, wm, wn, lane, acc);
            __syncthreads();
        }
    }

    // ---- epilogue: C layout -> row-major f32 ----
    const int crow = (lane >> 4) * 8;
    const int ccol = lane & 15;
    const int colBase = wn * 64;
#pragma unroll
    for (int m = 0; m < 4; ++m)
#pragma unroll
        for (int n = 0; n < 4; ++n)
#pragma unroll
            for (int v = 0; v < 8; ++v)
                Out[(size_t)(rowBase + wm * 64 + m * 16 + v + crow) * NOUT +
                    colBase + n * 16 + ccol] = acc[m][n][v];
}

// ---------------------------------------------------------------------------
extern "C" void kernel_launch(void* const* d_in, const int* in_sizes, int n_in,
                              void* d_out, int out_size, void* d_ws, size_t ws_size,
                              hipStream_t stream) {
    const float* X   = (const float*)d_in[0];   // [16384, 512]
    const float* bw1 = (const float*)d_in[1];   // [256, 512]
    const float* sw1 = (const float*)d_in[2];   // [256, 512, 8]
    const float* sc1 = (const float*)d_in[3];   // [256, 512]
    const float* bw2 = (const float*)d_in[4];   // [256, 256]
    const float* sw2 = (const float*)d_in[5];   // [256, 256, 8]
    const float* sc2 = (const float*)d_in[6];   // [256, 256]
    float* out = (float*)d_out;                 // [16384, 256]

    const int B = 16384, IN1 = 512, IN2 = 256;

    // workspace layout
    char* ws = (char*)d_ws;
    float*     x1  = (float*)ws;                               // 16 MB intermediate
    _Float16*  w1b = (_Float16*)(ws + (size_t)B * NOUT * 4);   // 512*256 f16
    _Float16*  w1s = w1b + (size_t)IN1 * NOUT;                 // 4096*256 f16
    _Float16*  w2b = w1s + (size_t)8 * IN1 * NOUT;             // 256*256 f16
    _Float16*  w2s = w2b + (size_t)IN2 * NOUT;                 // 2048*256 f16

    // pack weights into WMMA-B fragment layout (f16)
    pack_base_w  <<<(IN1 * NOUT) / NTHR, NTHR, 0, stream>>>(bw1, w1b, IN1);
    pack_spline_w<<<(8 * IN1 * NOUT) / NTHR, NTHR, 0, stream>>>(sw1, sc1, w1s, IN1);
    pack_base_w  <<<(IN2 * NOUT) / NTHR, NTHR, 0, stream>>>(bw2, w2b, IN2);
    pack_spline_w<<<(8 * IN2 * NOUT) / NTHR, NTHR, 0, stream>>>(sw2, sc2, w2s, IN2);

    // layer 1: [16384,512] -> [16384,256]
    kan_layer_gemm<<<B / BM, NTHR, 0, stream>>>(X, w1b, w1s, x1, IN1);
    // layer 2: [16384,256] -> [16384,256]
    kan_layer_gemm<<<B / BM, NTHR, 0, stream>>>(x1, w2b, w2s, out, IN2);
}
